// GMC_Att_25804163514497
// MI455X (gfx1250) — matrix operand order repaired and checked by reading, not compiled, not run on hardware
//
#include <hip/hip_runtime.h>

// ===========================================================================
// GMC attention block for MI455X (gfx1250, wave32, WMMA f32_16x16x32_f16).
//
// Assumptions:
//  * d_in order = setup_inputs() dict insertion order:
//      0:x  1:x_enhance_pre  2..63: params in _make_params insertion order
//      64: ref_idx (int32, single element, read on-device)
//  * ws_size >= ~290 MB
//
// Round 4: 2x2 register blocking in the GEMM kernels (32x32 tile per wave,
// 4 accumulators, 1 fragment-load per wmma). All GEMM operands are f16
// "panels" [kt][row][32]; every fragment load is 2x global_load_b128.
// ===========================================================================

typedef __attribute__((ext_vector_type(16))) _Float16 v16h;
typedef __attribute__((ext_vector_type(8)))  _Float16 v8h;
typedef __attribute__((ext_vector_type(8)))  float    v8f;

#define HW2 4096            // 64*64 spatial positions per frame
#define PF  (4096 * 32)     // panel elements per (frame, k-tile)

struct TDesc {
  const float* p;
  long long sb, st, rs, cs;  // batch stride, time stride, ref-idx stride, channel stride
  int div;                   // frame f -> (f/div)*sb + (f%div)*st + ref*rs
};

__device__ __forceinline__ const float* tbase(const TDesc& t, int f, int ref) {
  return t.p + (long long)(f / t.div) * t.sb + (long long)(f % t.div) * t.st +
         (long long)ref * t.rs;
}

__device__ __forceinline__ v8f zero8() {
  v8f z = {0.f, 0.f, 0.f, 0.f, 0.f, 0.f, 0.f, 0.f};
  return z;
}

__device__ __forceinline__ v8f wmma32(v16h a, v16h b, v8f c) {
  return __builtin_amdgcn_wmma_f32_16x16x32_f16(false, a, false, b, (short)0, c,
                                                false, false);
}

__device__ __forceinline__ v16h cat8(v8h u0, v8h u1) {
  return __builtin_shufflevector(u0, u1, 0, 1, 2, 3, 4, 5, 6, 7, 8, 9, 10, 11, 12,
                                 13, 14, 15);
}

// ---------------------------------------------------------------------------
// Panel fragment loaders. Panel layout: [kt][N rows][32 f16] (64B per row).
// A 16x32: row = lane&15; lane<16: K 0..15, lane>=16: K 16..31
//   -> two 16B runs per lane.
// B 32x16: col = lane&15; lane<16: K 0..15, lane>=16: K 16..31
//   -> one 32B run per lane (2x b128).
// ---------------------------------------------------------------------------
__device__ __forceinline__ v16h load_a_p(const _Float16* pan, int N, int kt,
                                         int row0, int lane) {
  int r = lane & 15;
  int kb = (lane < 16) ? 0 : 16;
  const _Float16* src = pan + ((long long)kt * N + row0 + r) * 32 + kb;
  v8h u0 = *(const v8h*)(src);       // k = kb .. kb+7
  v8h u1 = *(const v8h*)(src + 8);   // k = kb+8 .. kb+15
  return cat8(u0, u1);
}

__device__ __forceinline__ v16h load_b_p(const _Float16* pan, int N, int kt,
                                         int col0, int lane) {
  int n = lane & 15;
  int kb = (lane < 16) ? 0 : 16;
  const _Float16* src = pan + ((long long)kt * N + col0 + n) * 32 + kb;
  v8h u0 = *(const v8h*)(src);
  v8h u1 = *(const v8h*)(src + 8);
  return cat8(u0, u1);
}

// B fragment from f32 with contiguous K (for V operands): 4x b128
__device__ __forceinline__ v16h load_b_k1(const float* p, int ns, int lane) {
  int n = lane & 15;
  int kb = (lane < 16) ? 0 : 16;
  const float4* s4 = (const float4*)(p + n * ns + kb);
  float4 u0 = s4[0], u1 = s4[1], u2 = s4[2], u3 = s4[3];
  v16h b;
  b[0] = (_Float16)u0.x; b[1] = (_Float16)u0.y; b[2] = (_Float16)u0.z; b[3] = (_Float16)u0.w;
  b[4] = (_Float16)u1.x; b[5] = (_Float16)u1.y; b[6] = (_Float16)u1.z; b[7] = (_Float16)u1.w;
  b[8] = (_Float16)u2.x; b[9] = (_Float16)u2.y; b[10] = (_Float16)u2.z; b[11] = (_Float16)u2.w;
  b[12] = (_Float16)u3.x; b[13] = (_Float16)u3.y; b[14] = (_Float16)u3.z; b[15] = (_Float16)u3.w;
  return b;
}

// ---------------------------------------------------------------------------
// Pack strided f32 activation view -> f16 panel [f][KT][4096][32] (zero-pad K)
// grid ((4096*32)/256, KT, F), block 256. Coalesced reads, b16 writes.
// ---------------------------------------------------------------------------
__global__ void k_pack_act(TDesc in, const int* __restrict__ refp,
                           _Float16* __restrict__ outp, int C, int KT) {
  int id = blockIdx.x * 256 + threadIdx.x;  // [0, 131072)
  int kt = blockIdx.y;
  int f = blockIdx.z;
  int kk = id >> 12;        // 0..31
  int n = id & 4095;        // coalesced over n
  int c = kt * 32 + kk;
  int ref = refp ? refp[0] : 0;
  const float* inb = tbase(in, f, ref);
  float v = (c < C) ? inb[(long long)c * (int)in.cs + n] : 0.f;
  outp[((long long)f * KT + kt) * PF + (long long)n * 32 + kk] = (_Float16)v;
}

// ---------------------------------------------------------------------------
// Pack conv weight (Cout, Cin, taps) f32 -> panel [tap][KT][Mpad][32] f16,
// zero-padded in both K and M (M padded to 32). grid (Mpad*32/256, KT, ntaps)
// ---------------------------------------------------------------------------
__global__ void k_pack_w(const float* __restrict__ W, _Float16* __restrict__ outp,
                         int Cout, int Cin, int Mpad, int KT, int ntaps) {
  int id = blockIdx.x * 256 + threadIdx.x;  // [0, Mpad*32)
  int kt = blockIdx.y;
  int tap = blockIdx.z;
  int m = id >> 5, kk = id & 31;
  int k = kt * 32 + kk;
  float v = (m < Cout && k < Cin) ? W[((long long)m * Cin + k) * ntaps + tap] : 0.f;
  outp[(((long long)tap * KT + kt) * Mpad + m) * 32 + kk] = (_Float16)v;
}

// ---------------------------------------------------------------------------
// 1x1 conv = GEMM on panels, 2x2 register-blocked: 32(M)x32(N) per wave.
// epi: (+bias) -> (BN: *g*rsqrt(1+1e-5)+b) -> relu -> (*mask) -> (+resid)
// Output either f32 strided (outd) or f16 panel (outP, Cout multiple of 32).
// block 128 (4 waves); grid (32, Mpad/32, F)
// ---------------------------------------------------------------------------
__global__ void k_conv1x1(const _Float16* __restrict__ inp,
                          const _Float16* __restrict__ Wp, TDesc outd,
                          _Float16* __restrict__ outP, const float* __restrict__ bias,
                          const float* __restrict__ bng, const float* __restrict__ bnb,
                          const float* __restrict__ maskp, TDesc resid,
                          const int* __restrict__ refp, int KT, int Mpad, int MT,
                          int Cout, int has_resid, int relu) {
  int lane = threadIdx.x & 31;
  int wave = threadIdx.x >> 5;
  int n0 = (blockIdx.x * 4 + wave) * 32;
  int m0 = blockIdx.y * 32;
  int f = blockIdx.z;
  const _Float16* inf = inp + (long long)f * KT * PF;
  v8f acc[2][2];
#pragma unroll
  for (int i = 0; i < 2; ++i)
#pragma unroll
    for (int j = 0; j < 2; ++j) acc[i][j] = zero8();
  for (int kt = 0; kt < KT; ++kt) {
    v16h a0 = load_a_p(Wp, Mpad, kt, m0, lane);
    v16h a1 = load_a_p(Wp, Mpad, kt, m0 + 16, lane);
    v16h b0 = load_b_p(inf, HW2, kt, n0, lane);
    v16h b1 = load_b_p(inf, HW2, kt, n0 + 16, lane);
    acc[0][0] = wmma32(a0, b0, acc[0][0]);
    acc[0][1] = wmma32(a0, b1, acc[0][1]);
    acc[1][0] = wmma32(a1, b0, acc[1][0]);
    acc[1][1] = wmma32(a1, b1, acc[1][1]);
  }
  int col = lane & 15;
  int rbase = (lane < 16) ? 0 : 8;
  if (outP) {
#pragma unroll
    for (int i = 0; i < 2; ++i) {
      int mf = m0 + 16 * i + rbase;
#pragma unroll
      for (int jn = 0; jn < 2; ++jn) {
        v8h h;
#pragma unroll
        for (int r = 0; r < 8; ++r) {
          int m = mf + r;
          float v = acc[i][jn][r];
          if (bias) v += bias[m];
          if (bng) v = v * (bng[m] * rsqrtf(1.0f + 1e-5f)) + bnb[m];
          if (relu) v = fmaxf(v, 0.f);
          h[r] = (_Float16)v;
        }
        *(v8h*)(outP + (((long long)f * MT + (mf >> 5)) * HW2 +
                        (n0 + 16 * jn + col)) * 32 + (mf & 31)) = h;
      }
    }
  } else {
    int ref = refp ? refp[0] : 0;
    float* ob = (float*)tbase(outd, f, ref);
    const float* rb = has_resid ? tbase(resid, f, ref) : nullptr;
    int ocs = (int)outd.cs, rcs = (int)resid.cs;
#pragma unroll
    for (int i = 0; i < 2; ++i)
#pragma unroll
      for (int jn = 0; jn < 2; ++jn) {
        int nn = n0 + 16 * jn + col;
#pragma unroll
        for (int r = 0; r < 8; ++r) {
          int m = m0 + 16 * i + rbase + r;
          if (m < Cout) {
            float v = acc[i][jn][r];
            if (bias) v += bias[m];
            if (bng) v = v * (bng[m] * rsqrtf(1.0f + 1e-5f)) + bnb[m];
            if (relu) v = fmaxf(v, 0.f);
            if (maskp) v *= maskp[f * HW2 + nn];
            if (rb) v += rb[m * rcs + nn];
            ob[m * ocs + nn] = v;
          }
        }
      }
  }
}

// ---------------------------------------------------------------------------
// 3x3 SAME conv as implicit GEMM on panels, 2x2 register-blocked:
// 32 channels x 32 pixels (one row segment) per wave.
// Weights [tap][KT][Mpad][32]. Output f32 plain NCHW.
// block 128; grid (32, Mpad/32, F)
// ---------------------------------------------------------------------------
__global__ void k_conv3x3(const _Float16* __restrict__ inp,
                          const _Float16* __restrict__ Wp,
                          const float* __restrict__ bias, float* __restrict__ out,
                          int KT, int Mpad, int Cout) {
  int lane = threadIdx.x & 31;
  int wave = threadIdx.x >> 5;
  int tile = blockIdx.x * 4 + wave;  // 128 tiles of 32 pixels
  int y = tile >> 1;
  int x0 = (tile & 1) * 32;
  int m0 = blockIdx.y * 32;
  int f = blockIdx.z;
  const _Float16* inf = inp + (long long)f * KT * PF;
  long long tapstride = (long long)KT * Mpad * 32;
  int n = lane & 15;
  int kbB = (lane < 16) ? 0 : 16;
  v16h bz;
#pragma unroll
  for (int j = 0; j < 16; ++j) bz[j] = (_Float16)0.f;
  v8f acc[2][2];
#pragma unroll
  for (int i = 0; i < 2; ++i)
#pragma unroll
    for (int j = 0; j < 2; ++j) acc[i][j] = zero8();
  for (int kt = 0; kt < KT; ++kt) {
#pragma unroll
    for (int dy = -1; dy <= 1; ++dy) {
      int yy = y + dy;
      bool yok = ((unsigned)yy < 64u);
#pragma unroll
      for (int dx = -1; dx <= 1; ++dx) {
        int tap = (dy + 1) * 3 + (dx + 1);
        v16h a0 = load_a_p(Wp + tap * tapstride, Mpad, kt, m0, lane);
        v16h a1 = load_a_p(Wp + tap * tapstride, Mpad, kt, m0 + 16, lane);
        v16h b[2];
#pragma unroll
        for (int seg = 0; seg < 2; ++seg) {
          int xx = x0 + 16 * seg + n + dx;
          bool ok = yok && ((unsigned)xx < 64u);
          const _Float16* src =
              inf + ((long long)kt * HW2 + (ok ? (yy * 64 + xx) : 0)) * 32 + kbB;
          v8h u0 = *(const v8h*)src;
          v8h u1 = *(const v8h*)(src + 8);
          v16h bb = cat8(u0, u1);
          b[seg] = ok ? bb : bz;
        }
        acc[0][0] = wmma32(a0, b[0], acc[0][0]);
        acc[0][1] = wmma32(a0, b[1], acc[0][1]);
        acc[1][0] = wmma32(a1, b[0], acc[1][0]);
        acc[1][1] = wmma32(a1, b[1], acc[1][1]);
      }
    }
  }
  int col = lane & 15;
  int rbase = (lane < 16) ? 0 : 8;
  float* ob = out + (long long)f * Cout * HW2 + y * 64 + x0 + col;
#pragma unroll
  for (int i = 0; i < 2; ++i)
#pragma unroll
    for (int jn = 0; jn < 2; ++jn)
#pragma unroll
      for (int r = 0; r < 8; ++r) {
        int m = m0 + 16 * i + rbase + r;
        if (m < Cout) ob[m * HW2 + 16 * jn] = acc[i][jn][r] + (bias ? bias[m] : 0.f);
      }
}

// ---------------------------------------------------------------------------
// Depthwise 3x3 conv (+bias, optional BN+ReLU). thread per pixel.
// Writes f32 plain and/or f16 panel. grid (16, C, F), block 256
// ---------------------------------------------------------------------------
__global__ void k_dw(const float* __restrict__ in, const float* __restrict__ W,
                     const float* __restrict__ bias, const float* __restrict__ bng,
                     const float* __restrict__ bnb, float* __restrict__ outF,
                     _Float16* __restrict__ outP, int C, int relu) {
  int n = blockIdx.x * blockDim.x + threadIdx.x;
  int c = blockIdx.y;
  int f = blockIdx.z;
  int y = n >> 6, x = n & 63;
  const float* p = in + ((long long)f * C + c) * HW2;
  const float* w = W + c * 9;
  float s = bias ? bias[c] : 0.f;
#pragma unroll
  for (int dy = -1; dy <= 1; ++dy) {
    int yy = y + dy;
    if ((unsigned)yy >= 64u) continue;
#pragma unroll
    for (int dx = -1; dx <= 1; ++dx) {
      int xx = x + dx;
      if ((unsigned)xx >= 64u) continue;
      s += w[(dy + 1) * 3 + (dx + 1)] * p[yy * 64 + xx];
    }
  }
  if (bng) s = s * (bng[c] * rsqrtf(1.0f + 1e-5f)) + bnb[c];
  if (relu) s = fmaxf(s, 0.f);
  if (outF) outF[((long long)f * C + c) * HW2 + n] = s;
  if (outP)
    outP[(((long long)f * (C >> 5) + (c >> 5)) * HW2 + n) * 32 + (c & 31)] =
        (_Float16)s;
}

// ---------------------------------------------------------------------------
// Channel LayerNorm (NCHW over C) + optional ReLU; writes f16 panel with
// zero-padded K (KT*32 >= C). One wave per pixel. grid (512, F), block 256.
// ---------------------------------------------------------------------------
__global__ void k_ln(const float* __restrict__ in, const float* __restrict__ w,
                     const float* __restrict__ b, _Float16* __restrict__ outP, int C,
                     int KT, int relu) {
  int lane = threadIdx.x & 31;
  int wave = threadIdx.x >> 5;
  int n = blockIdx.x * 8 + wave;
  int f = blockIdx.y;
  const float* p = in + (long long)f * C * HW2 + n;
  float s = 0.f, s2 = 0.f;
  for (int c = lane; c < C; c += 32) {
    float v = p[c * HW2];
    s += v;
    s2 += v * v;
  }
#pragma unroll
  for (int off = 16; off > 0; off >>= 1) {
    s += __shfl_xor(s, off, 32);
    s2 += __shfl_xor(s2, off, 32);
  }
  float mu = s / C;
  float inv = rsqrtf(s2 / C - mu * mu + 1e-5f);
  int Cpad = KT * 32;
  for (int c = lane; c < Cpad; c += 32) {
    int cc = (c < C) ? c : 0;
    float v = (p[cc * HW2] - mu) * inv * w[cc] + b[cc];
    if (relu) v = fmaxf(v, 0.f);
    if (c >= C) v = 0.f;
    outP[(((long long)f * KT + (c >> 5)) * HW2 + n) * 32 + (c & 31)] = (_Float16)v;
  }
}

// ---------------------------------------------------------------------------
// Attention pass 1: online-softmax stats. scores = (Q . Kk) / 8.
// qP: panels (8 frames, KT=2), kkP: panels (2 frames, KT=2).
// One wave = 16 query rows. grid (64, 8), block 128
// ---------------------------------------------------------------------------
__global__ void k_attn_stats(const _Float16* __restrict__ qP,
                             const _Float16* __restrict__ kkP,
                             float* __restrict__ smax, float* __restrict__ zsum) {
  int lane = threadIdx.x & 31;
  int wave = threadIdx.x >> 5;
  int f = blockIdx.y;
  int nq0 = (blockIdx.x * 4 + wave) * 16;
  const _Float16* qf = qP + (long long)f * 2 * PF;
  const _Float16* kkf = kkP + (long long)(f >> 2) * 2 * PF;
  v16h a0 = load_a_p(qf, HW2, 0, nq0, lane);
  v16h a1 = load_a_p(qf, HW2, 1, nq0, lane);
  float m[8], z[8];
#pragma unroll
  for (int r = 0; r < 8; ++r) { m[r] = -1e30f; z[r] = 0.f; }
  for (int j0 = 0; j0 < HW2; j0 += 16) {
    v8f acc = zero8();
    v16h b0 = load_b_p(kkf, HW2, 0, j0, lane);
    v16h b1 = load_b_p(kkf, HW2, 1, j0, lane);
    acc = wmma32(a0, b0, acc);
    acc = wmma32(a1, b1, acc);
#pragma unroll
    for (int r = 0; r < 8; ++r) {
      float sline = acc[r] * 0.125f;
      float tm = sline;
#pragma unroll
      for (int w = 8; w > 0; w >>= 1) tm = fmaxf(tm, __shfl_xor(tm, w, 32));
      float mn = fmaxf(m[r], tm);
      float e = __expf(sline - mn);
#pragma unroll
      for (int w = 8; w > 0; w >>= 1) e += __shfl_xor(e, w, 32);
      z[r] = z[r] * __expf(m[r] - mn) + e;
      m[r] = mn;
    }
  }
  if ((lane & 15) == 0) {
    int rb = (lane < 16) ? 0 : 8;
    long long base = (long long)f * HW2 + nq0 + rb;
#pragma unroll
    for (int r = 0; r < 8; ++r) {
      smax[base + r] = m[r];
      zsum[base + r] = z[r];
    }
  }
}

// ---------------------------------------------------------------------------
// Attention pass 2: feat[f, c, nq] = (1/Z) sum_m exp(s-smax) * kv[c, m]
// P tiles bounce through LDS (D-layout -> A-layout). grid (64, 8), block 128
// ---------------------------------------------------------------------------
__global__ void k_attn_av(const _Float16* __restrict__ qP,
                          const _Float16* __restrict__ kkP,
                          const float* __restrict__ kv, const float* __restrict__ smax,
                          const float* __restrict__ zsum, float* __restrict__ feat) {
  __shared__ _Float16 lp[4][16 * 32];
  int lane = threadIdx.x & 31;
  int wave = threadIdx.x >> 5;
  int f = blockIdx.y;
  int nq0 = (blockIdx.x * 4 + wave) * 16;
  const _Float16* qf = qP + (long long)f * 2 * PF;
  const _Float16* kkf = kkP + (long long)(f >> 2) * 2 * PF;
  const float* vb = kv + (long long)(f >> 2) * 64 * HW2;
  v16h a0 = load_a_p(qf, HW2, 0, nq0, lane);
  v16h a1 = load_a_p(qf, HW2, 1, nq0, lane);
  int col = lane & 15;
  int rbase = (lane < 16) ? 0 : 8;
  float sm[8], zz[8];
#pragma unroll
  for (int r = 0; r < 8; ++r) {
    long long i = (long long)f * HW2 + nq0 + rbase + r;
    sm[r] = smax[i];
    zz[r] = zsum[i];
  }
  v8f acc0 = zero8(), acc1 = zero8(), acc2 = zero8(), acc3 = zero8();
  for (int m0 = 0; m0 < HW2; m0 += 32) {
#pragma unroll
    for (int h = 0; h < 2; ++h) {
      v8f s = zero8();
      v16h b0 = load_b_p(kkf, HW2, 0, m0 + 16 * h, lane);
      v16h b1 = load_b_p(kkf, HW2, 1, m0 + 16 * h, lane);
      s = wmma32(a0, b0, s);
      s = wmma32(a1, b1, s);
#pragma unroll
      for (int r = 0; r < 8; ++r)
        lp[wave][(rbase + r) * 32 + h * 16 + col] =
            (_Float16)__expf(s[r] * 0.125f - sm[r]);
    }
    v16h pa;
    {
      int rr = lane & 15;
      int kb2 = (lane < 16) ? 0 : 16;
#pragma unroll
      for (int j = 0; j < 16; ++j) {
        int k = j + kb2;
        pa[j] = lp[wave][rr * 32 + k];
      }
    }
    acc0 = wmma32(pa, load_b_k1(vb + 0LL * 16 * HW2 + m0, HW2, lane), acc0);
    acc1 = wmma32(pa, load_b_k1(vb + 1LL * 16 * HW2 + m0, HW2, lane), acc1);
    acc2 = wmma32(pa, load_b_k1(vb + 2LL * 16 * HW2 + m0, HW2, lane), acc2);
    acc3 = wmma32(pa, load_b_k1(vb + 3LL * 16 * HW2 + m0, HW2, lane), acc3);
  }
  float* ob = feat + (long long)f * 64 * HW2;
#pragma unroll
  for (int r = 0; r < 8; ++r) {
    int nq = nq0 + rbase + r;
    float invz = 1.f / zz[r];
    ob[(0 * 16 + col) * HW2 + nq] = acc0[r] * invz;
    ob[(1 * 16 + col) * HW2 + nq] = acc1[r] * invz;
    ob[(2 * 16 + col) * HW2 + nq] = acc2[r] * invz;
    ob[(3 * 16 + col) * HW2 + nq] = acc3[r] * invz;
  }
}

// ---------------------------------------------------------------------------
// Attention pass 3: featk[b, c, m] = (1/Tm) sum_t sum_n P[t,n,m] * v[t, c, n]
// (P^T . V via S^T tiles). grid (64, 2), block 128
// ---------------------------------------------------------------------------
__global__ void k_attn_key(const _Float16* __restrict__ qP,
                           const _Float16* __restrict__ kkP,
                           const float* __restrict__ vf, const float* __restrict__ smax,
                           const float* __restrict__ zsum, float* __restrict__ featk) {
  __shared__ _Float16 lp[4][16 * 32];
  int lane = threadIdx.x & 31;
  int wave = threadIdx.x >> 5;
  int b = blockIdx.y;
  int m0 = (blockIdx.x * 4 + wave) * 16;
  const _Float16* kkf = kkP + (long long)b * 2 * PF;
  v16h a0 = load_a_p(kkf, HW2, 0, m0, lane);
  v16h a1 = load_a_p(kkf, HW2, 1, m0, lane);
  int col = lane & 15;
  int rbase = (lane < 16) ? 0 : 8;
  v8f acc0 = zero8(), acc1 = zero8(), acc2 = zero8(), acc3 = zero8();
  for (int t = 0; t < 4; ++t) {
    int f = b * 4 + t;
    const _Float16* qf = qP + (long long)f * 2 * PF;
    const float* vb = vf + (long long)f * 64 * HW2;
    long long sbase = (long long)f * HW2;
    for (int n0 = 0; n0 < HW2; n0 += 32) {
#pragma unroll
      for (int h = 0; h < 2; ++h) {
        v8f s = zero8();
        v16h b0 = load_b_p(qf, HW2, 0, n0 + 16 * h, lane);
        v16h b1 = load_b_p(qf, HW2, 1, n0 + 16 * h, lane);
        s = wmma32(a0, b0, s);
        s = wmma32(a1, b1, s);
        float smv = smax[sbase + n0 + 16 * h + col];
        float izv = 1.f / zsum[sbase + n0 + 16 * h + col];
#pragma unroll
        for (int r = 0; r < 8; ++r)
          lp[wave][(rbase + r) * 32 + h * 16 + col] =
              (_Float16)(__expf(s[r] * 0.125f - smv) * izv);
      }
      v16h pa;
      {
        int rr = lane & 15;
        int kb2 = (lane < 16) ? 0 : 16;
#pragma unroll
        for (int j = 0; j < 16; ++j) {
          int k = j + kb2;
          pa[j] = lp[wave][rr * 32 + k];
        }
      }
      acc0 = wmma32(pa, load_b_k1(vb + 0LL * 16 * HW2 + n0, HW2, lane), acc0);
      acc1 = wmma32(pa, load_b_k1(vb + 1LL * 16 * HW2 + n0, HW2, lane), acc1);
      acc2 = wmma32(pa, load_b_k1(vb + 2LL * 16 * HW2 + n0, HW2, lane), acc2);
      acc3 = wmma32(pa, load_b_k1(vb + 3LL * 16 * HW2 + n0, HW2, lane), acc3);
    }
  }
  float* ob = featk + (long long)b * 64 * HW2;
#pragma unroll
  for (int r = 0; r < 8; ++r) {
    int mrow = m0 + rbase + r;
    ob[(0 * 16 + col) * HW2 + mrow] = acc0[r] * 0.25f;
    ob[(1 * 16 + col) * HW2 + mrow] = acc1[r] * 0.25f;
    ob[(2 * 16 + col) * HW2 + mrow] = acc2[r] * 0.25f;
    ob[(3 * 16 + col) * HW2 + mrow] = acc3[r] * 0.25f;
  }
}

// ---------------------------------------------------------------------------
// mask = sigmoid(conv3x3([avg=1/4096, mx=1/Z], mask_w) + mask_b)
// grid (16, 8), block 256
// ---------------------------------------------------------------------------
__global__ void k_mask(const float* __restrict__ zsum, const float* __restrict__ mw,
                       const float* __restrict__ mb, float* __restrict__ mask) {
  int n = blockIdx.x * blockDim.x + threadIdx.x;
  int f = blockIdx.y;
  int y = n >> 6, x = n & 63;
  const float* zf = zsum + (long long)f * HW2;
  float s = mb[0];
#pragma unroll
  for (int dy = -1; dy <= 1; ++dy) {
    int yy = y + dy;
    if ((unsigned)yy >= 64u) continue;
#pragma unroll
    for (int dx = -1; dx <= 1; ++dx) {
      int xx = x + dx;
      if ((unsigned)xx >= 64u) continue;
      int tap = (dy + 1) * 3 + (dx + 1);
      s += mw[tap] * (1.f / 4096.f);
      s += mw[9 + tap] * (1.f / zf[yy * 64 + xx]);
    }
  }
  mask[(long long)f * HW2 + n] = 1.f / (1.f + __expf(-s));
}

// ===========================================================================
extern "C" void kernel_launch(void* const* d_in, const int* in_sizes, int n_in,
                              void* d_out, int out_size, void* d_ws, size_t ws_size,
                              hipStream_t stream) {
  (void)in_sizes; (void)out_size; (void)ws_size;
  const float* x = (const float*)d_in[0];
  const float* P[64];
  for (int i = 0; i < 64 && i < n_in; ++i) P[i] = (const float*)d_in[i];
  const int* refp = (n_in > 64) ? (const int*)d_in[64] : nullptr;
  float* out = (float*)d_out;

  // byte bump-allocator over workspace
  size_t off = 0;
  auto allocB = [&](size_t bytes) {
    void* r = (char*)d_ws + off;
    off += (bytes + 255) & ~(size_t)255;
    return r;
  };
  auto aF = [&](size_t n) { return (float*)allocB(n * 4); };
  auto aH = [&](size_t n) { return (_Float16*)allocB(n * 2); };

  // f32 buffers
  float* ef8   = aF(8ull * 64 * HW2);   // emb2 out (8f) -> later feat (pass2 out)
  float* efK   = aF(2ull * 64 * HW2);   // emb2 out (key frames)
  float* vfb   = aF(8ull * 64 * HW2);   // refv
  float* keyv  = aF(2ull * 64 * HW2);
  float* h340  = aF(8ull * 340 * HW2);
  float* feat1 = aF(8ull * 64 * HW2);
  float* big2  = aF(8ull * 512 * HW2);
  float* featk = aF(2ull * 64 * HW2);
  float* kfeat1= aF(2ull * 64 * HW2);
  float* smaxb = aF(8ull * HW2);
  float* zsumb = aF(8ull * HW2);
  float* maskb = aF(8ull * HW2);

  // f16 activation panels
  _Float16* xP8  = aH(8ull * 16 * PF);
  _Float16* xPK  = aH(2ull * 16 * PF);
  _Float16* e1P8 = aH(8ull * 2 * PF);
  _Float16* e1PK = aH(2ull * 2 * PF);
  _Float16* dwP8 = aH(8ull * 2 * PF);
  _Float16* dwPK = aH(2ull * 2 * PF);
  _Float16* qfP  = aH(8ull * 2 * PF);
  _Float16* kkP  = aH(2ull * 2 * PF);
  _Float16* lnP8 = aH(8ull * 2 * PF);   // ln1 out, then proj1-LN out (reused)
  _Float16* h340P= aH(8ull * 11 * PF);
  _Float16* big1P= aH(8ull * 16 * PF);  // c1 out -> (reused) dw out
  _Float16* klnP = aH(2ull * 2 * PF);   // key-branch LN panels (reused)

  // f16 weight panels
  _Float16* wEmb1 = aH(16ull * 64 * 32);
  _Float16* wEmb2 = aH(9ull * 2 * 64 * 32);
  _Float16* wQ    = aH(2ull * 64 * 32);
  _Float16* wV    = aH(2ull * 64 * 32);
  _Float16* wKK   = aH(2ull * 64 * 32);
  _Float16* wKV   = aH(2ull * 64 * 32);
  _Float16* wF1i  = aH(2ull * 352 * 32);
  _Float16* wF1o  = aH(11ull * 64 * 32);
  _Float16* wF2i  = aH(2ull * 352 * 32);
  _Float16* wF2o  = aH(11ull * 64 * 32);
  _Float16* wP1c1 = aH(2ull * 512 * 32);
  _Float16* wP1c2 = aH(9ull * 16 * 512 * 32);
  _Float16* wP2c1 = aH(2ull * 512 * 32);
  _Float16* wP2c2 = aH(9ull * 16 * 512 * 32);
  _Float16* wOref = aH(16ull * 512 * 32);
  _Float16* wOkey = aH(16ull * 512 * 32);

  auto plain = [](const float* p, int C) {
    TDesc t{p, 0, (long long)C * HW2, 0, HW2, 1 << 28};
    return t;
  };
  const long long XB = 512ll * 5 * HW2;
  TDesc xfD{x, XB, HW2, 0, 5ll * HW2, 4};              // xf frames f=b*4+t
  TDesc xkeyD{x, 0, XB, HW2, 5ll * HW2, 1 << 28};      // key frames (uses ref_idx)
  TDesc orefD{out, XB, HW2, 0, 5ll * HW2, 4};          // out[:, :, t]
  TDesc okeyD{out + 4ll * HW2, 0, XB, 0, 5ll * HW2, 1 << 28};  // out[:, :, 4]
  TDesc dz{nullptr, 0, 0, 0, 0, 1};

  dim3 B128(128), B256(256);

  // ---- pack all conv weights into panels (M padded to 32) ----
  auto packw = [&](const float* W, _Float16* dst, int Cout, int Cin, int ntaps) {
    int Mpad = (Cout + 31) & ~31;
    int KT = (Cin + 31) / 32;
    k_pack_w<<<dim3(Mpad * 32 / 256, KT, ntaps), B256, 0, stream>>>(W, dst, Cout,
                                                                    Cin, Mpad, KT,
                                                                    ntaps);
  };
  packw(P[2], wEmb1, 64, 512, 1);
  packw(P[4], wEmb2, 64, 64, 9);
  packw(P[8], wQ, 64, 64, 1);
  packw(P[12], wV, 64, 64, 1);
  packw(P[16], wKK, 64, 64, 1);
  packw(P[20], wKV, 64, 64, 1);
  packw(P[26], wF1i, 340, 64, 1);
  packw(P[29], wF1o, 64, 340, 1);
  packw(P[30], wF2i, 340, 64, 1);
  packw(P[33], wF2o, 64, 340, 1);
  packw(P[36], wP1c1, 512, 64, 1);
  packw(P[40], wP1c2, 512, 512, 9);
  packw(P[44], wP2c1, 512, 64, 1);
  packw(P[48], wP2c2, 512, 512, 9);
  packw(P[54], wOref, 512, 512, 1);
  packw(P[60], wOkey, 512, 512, 1);

  // ---- pack input x into panels ----
  k_pack_act<<<dim3(512, 16, 8), B256, 0, stream>>>(xfD, nullptr, xP8, 512, 16);
  k_pack_act<<<dim3(512, 16, 2), B256, 0, stream>>>(xkeyD, refp, xPK, 512, 16);

  // ---- key-frame embedding + key K/V ----
  k_conv1x1<<<dim3(32, 2, 2), B128, 0, stream>>>(xPK, wEmb1, dz, e1PK, P[3], nullptr, nullptr, nullptr, dz, nullptr, 16, 64, 2, 64, 0, 0);
  k_conv3x3<<<dim3(32, 2, 2), B128, 0, stream>>>(e1PK, wEmb2, P[5], efK, 2, 64, 64);
  k_dw<<<dim3(16, 64, 2), B256, 0, stream>>>(efK, P[14], P[15], nullptr, nullptr, nullptr, dwPK, 64, 0);
  k_conv1x1<<<dim3(32, 2, 2), B128, 0, stream>>>(dwPK, wKK, dz, kkP, P[17], nullptr, nullptr, nullptr, dz, nullptr, 2, 64, 2, 64, 0, 0);
  k_dw<<<dim3(16, 64, 2), B256, 0, stream>>>(efK, P[18], P[19], nullptr, nullptr, nullptr, dwPK, 64, 0);
  k_conv1x1<<<dim3(32, 2, 2), B128, 0, stream>>>(dwPK, wKV, plain(keyv, 64), nullptr, P[21], nullptr, nullptr, nullptr, dz, nullptr, 2, 64, 0, 64, 0, 0);

  // ---- 8-frame embedding + Q / V ----
  k_conv1x1<<<dim3(32, 2, 8), B128, 0, stream>>>(xP8, wEmb1, dz, e1P8, P[3], nullptr, nullptr, nullptr, dz, nullptr, 16, 64, 2, 64, 0, 0);
  k_conv3x3<<<dim3(32, 2, 8), B128, 0, stream>>>(e1P8, wEmb2, P[5], ef8, 2, 64, 64);
  k_dw<<<dim3(16, 64, 8), B256, 0, stream>>>(ef8, P[6], P[7], nullptr, nullptr, nullptr, dwP8, 64, 0);
  k_conv1x1<<<dim3(32, 2, 8), B128, 0, stream>>>(dwP8, wQ, dz, qfP, P[9], nullptr, nullptr, nullptr, dz, nullptr, 2, 64, 2, 64, 0, 0);
  k_dw<<<dim3(16, 64, 8), B256, 0, stream>>>(ef8, P[10], P[11], nullptr, nullptr, nullptr, dwP8, 64, 0);
  k_conv1x1<<<dim3(32, 2, 8), B128, 0, stream>>>(dwP8, wV, plain(vfb, 64), nullptr, P[13], nullptr, nullptr, nullptr, dz, nullptr, 2, 64, 0, 64, 0, 0);

  // ---- attention (stats, enh, mask) ----
  k_attn_stats<<<dim3(64, 8), B128, 0, stream>>>(qfP, kkP, smaxb, zsumb);
  k_attn_av<<<dim3(64, 8), B128, 0, stream>>>(qfP, kkP, keyv, smaxb, zsumb, ef8);
  k_mask<<<dim3(16, 8), B256, 0, stream>>>(zsumb, P[62], P[63], maskb);

  // ---- ref branch: feat += ffn1(ln1(feat)); proj1; oref; *mask + xf ----
  k_ln<<<dim3(512, 8), B256, 0, stream>>>(ef8, P[22], P[23], lnP8, 64, 2, 0);
  k_conv1x1<<<dim3(32, 11, 8), B128, 0, stream>>>(lnP8, wF1i, plain(h340, 340), nullptr, nullptr, nullptr, nullptr, nullptr, dz, nullptr, 2, 352, 0, 340, 0, 0);
  k_ln<<<dim3(512, 8), B256, 0, stream>>>(h340, P[27], P[28], h340P, 340, 11, 1);
  k_conv1x1<<<dim3(32, 2, 8), B128, 0, stream>>>(h340P, wF1o, plain(feat1, 64), nullptr, nullptr, nullptr, nullptr, nullptr, plain(ef8, 64), nullptr, 11, 64, 0, 64, 1, 0);
  k_ln<<<dim3(512, 8), B256, 0, stream>>>(feat1, P[34], P[35], lnP8, 64, 2, 0);
  k_conv1x1<<<dim3(32, 16, 8), B128, 0, stream>>>(lnP8, wP1c1, dz, big1P, P[37], P[38], P[39], nullptr, dz, nullptr, 2, 512, 16, 512, 0, 1);
  k_conv3x3<<<dim3(32, 16, 8), B128, 0, stream>>>(big1P, wP1c2, P[41], big2, 16, 512, 512);
  k_dw<<<dim3(16, 512, 8), B256, 0, stream>>>(big2, P[50], P[51], P[52], P[53], nullptr, big1P, 512, 1);
  k_conv1x1<<<dim3(32, 16, 8), B128, 0, stream>>>(big1P, wOref, orefD, nullptr, P[55], nullptr, nullptr, maskb, xfD, nullptr, 16, 512, 0, 512, 1, 0);

  // ---- key branch: featk += ffn2(ln2(featk)); proj2; okey; + key_feat ----
  k_attn_key<<<dim3(64, 2), B128, 0, stream>>>(qfP, kkP, vfb, smaxb, zsumb, featk);
  k_ln<<<dim3(512, 2), B256, 0, stream>>>(featk, P[24], P[25], klnP, 64, 2, 0);
  k_conv1x1<<<dim3(32, 11, 2), B128, 0, stream>>>(klnP, wF2i, plain(h340, 340), nullptr, nullptr, nullptr, nullptr, nullptr, dz, nullptr, 2, 352, 0, 340, 0, 0);
  k_ln<<<dim3(512, 2), B256, 0, stream>>>(h340, P[31], P[32], h340P, 340, 11, 1);
  k_conv1x1<<<dim3(32, 2, 2), B128, 0, stream>>>(h340P, wF2o, plain(kfeat1, 64), nullptr, nullptr, nullptr, nullptr, nullptr, plain(featk, 64), nullptr, 11, 64, 0, 64, 1, 0);
  k_ln<<<dim3(512, 2), B256, 0, stream>>>(kfeat1, P[42], P[43], klnP, 64, 2, 0);
  k_conv1x1<<<dim3(32, 16, 2), B128, 0, stream>>>(klnP, wP2c1, dz, big1P, P[45], P[46], P[47], nullptr, dz, nullptr, 2, 512, 16, 512, 0, 1);
  k_conv3x3<<<dim3(32, 16, 2), B128, 0, stream>>>(big1P, wP2c2, P[49], big2, 16, 512, 512);
  k_dw<<<dim3(16, 512, 2), B256, 0, stream>>>(big2, P[56], P[57], P[58], P[59], nullptr, big1P, 512, 1);
  k_conv1x1<<<dim3(32, 16, 2), B128, 0, stream>>>(big1P, wOkey, okeyD, nullptr, P[61], nullptr, nullptr, nullptr, xkeyD, refp, 16, 512, 0, 512, 1, 0);
}
